// AAFlow_6983616824247
// MI455X (gfx1250) — compile-verified, matrix operand order checked by reading, not run on hardware
//
#include <hip/hip_runtime.h>
#include <hip/hip_bf16.h>

typedef _Float16 f16;
typedef __attribute__((ext_vector_type(16))) _Float16 v16h;
typedef __attribute__((ext_vector_type(8)))  _Float16 v8h;
typedef __attribute__((ext_vector_type(8)))  float    v8f;

#define Bsz 8
#define Nn  128
#define Hd  256
#define Vv  21
#define CRconst (15.0f / 4.0f)   // coords_range / n_layers = 15/4

__device__ __forceinline__ float silu_f(float x) {
    return x * (1.f / (1.f + __expf(-x)));
}

// ---- WMMA helpers -------------------------------------------------------
// A fragment (16x32 f16, row-major source in LDS with row stride `ld` halves).
// Lane L holds row m=L&15; halves 0..7 -> k = (L>>4)*8 + 0..7,
// halves 8..15 -> k = 16 + (L>>4)*8 + 0..7.
__device__ __forceinline__ v16h lds_a_frag(const f16* A, int mt, int kt, int lane, int ld) {
    int m  = mt * 16 + (lane & 15);
    int hi = lane >> 4;
    const f16* p = A + m * ld + kt * 32 + hi * 8;
    v8h lo = *reinterpret_cast<const v8h*>(p);
    v8h hp = *reinterpret_cast<const v8h*>(p + 16);
    v16h r;
#pragma unroll
    for (int t = 0; t < 8; ++t) { r[t] = lo[t]; r[t + 8] = hp[t]; }
    return r;
}

// ---- weight packing: row-major f32 W[K][N] -> B-fragment order f16 ------
// dst layout: ((kt*NT + nt)*32 + lane)*16 + h ; per-lane contiguous 32B.
__global__ void pack_wfrag(const float* __restrict__ w, int Ksrc, int Nsrc,
                           int NT, f16* __restrict__ dst, int total) {
    int tid = blockIdx.x * 256 + threadIdx.x;
    if (tid >= total) return;
    int h    = tid & 15;
    int lane = (tid >> 4) & 31;
    int ft   = tid >> 9;
    int nt   = ft % NT;
    int kt   = ft / NT;
    int n = nt * 16 + (lane & 15);
    int k = kt * 32 + (h & 7) + (lane >> 4) * 8 + ((h & 8) ? 16 : 0);
    float v = (k < Ksrc && n < Nsrc) ? w[(size_t)k * Nsrc + n] : 0.f;
    dst[tid] = (f16)v;
}

// ---- embedding: h0 = (atom_emb[type] + time_emb) * mask -----------------
__global__ void embed_kernel(const float* __restrict__ atom_emb, const int* __restrict__ types,
                             const float* __restrict__ maskv, const float* __restrict__ time_emb,
                             const int* __restrict__ tC, const int* __restrict__ tA,
                             float* __restrict__ h) {
    int row = blockIdx.x;           // 0..1023
    int c   = threadIdx.x;          // 0..255
    int b   = row >> 7;
    float te = (c < 128) ? time_emb[tC[b] * 128 + c]
                         : time_emb[tA[b] * 128 + (c - 128)];
    float m = maskv[row];
    h[(size_t)row * Hd + c] = (atom_emb[(size_t)types[row] * Hd + c] + te) * m;
}

// ---- generic node-level GEMM via WMMA -----------------------------------
// out = post( A[M,K] @ Bfrag + bias ), 32 rows / block, M = gridDim.x*32
// A source: A16 (f16 row-major) if non-null, else concat(A0[:,0:K0], A1[:,K0:K]).
// act: 0=none, 1=silu, 2=relu ; optional residual (+resid[row,256]) & mask (*maskv[row]).
__global__ void node_gemm(const float* __restrict__ A0, const float* __restrict__ A1,
                          const f16* __restrict__ A16, int K0, int K,
                          const f16* __restrict__ Bfrag, const float* __restrict__ bias,
                          int Npack, int Nwrite, int act,
                          const float* __restrict__ resid, const float* __restrict__ maskv,
                          float* __restrict__ outF, int outStride,
                          f16* __restrict__ outH) {
    __shared__ alignas(32) f16 Asm[32 * 520];
    const int ld   = K + 8;
    const int row0 = blockIdx.x * 32;
    const int tid  = threadIdx.x;
    for (int idx = tid; idx < 32 * K; idx += 256) {
        int r = idx / K, k = idx - r * K;
        float v;
        if (A16)         v = (float)A16[(size_t)(row0 + r) * K + k];
        else if (k < K0) v = A0[(size_t)(row0 + r) * K0 + k];
        else             v = A1[(size_t)(row0 + r) * (K - K0) + (k - K0)];
        Asm[r * ld + k] = (f16)v;
    }
    __syncthreads();
    const int lane = tid & 31, wave = tid >> 5;
    const int NT = Npack >> 4, KT = K >> 5;
    const v16h* Bv = reinterpret_cast<const v16h*>(Bfrag);

    if (NT == 16) {
        // 4-wide: wave owns (mt = w>>2) x 4 consecutive n-tiles.
        const int mt = wave >> 2;
        const int ntBase = (wave & 3) * 4;
        v8f c0 = {}, c1 = {}, c2 = {}, c3 = {};
        for (int kt = 0; kt < KT; ++kt) {
            v16h a = lds_a_frag(Asm, mt, kt, lane, ld);
            const v16h* bp = Bv + (size_t)(kt * 16 + ntBase) * 32 + lane;
            v16h b0 = bp[0];
            v16h b1 = bp[32];
            v16h b2 = bp[64];
            v16h b3 = bp[96];
            c0 = __builtin_amdgcn_wmma_f32_16x16x32_f16(false, a, false, b0, (short)0, c0, false, false);
            c1 = __builtin_amdgcn_wmma_f32_16x16x32_f16(false, a, false, b1, (short)0, c1, false, false);
            c2 = __builtin_amdgcn_wmma_f32_16x16x32_f16(false, a, false, b2, (short)0, c2, false, false);
            c3 = __builtin_amdgcn_wmma_f32_16x16x32_f16(false, a, false, b3, (short)0, c3, false, false);
        }
        const int mb = mt * 16 + (lane >> 4) * 8;
        v8f acc[4] = {c0, c1, c2, c3};
#pragma unroll
        for (int q = 0; q < 4; ++q) {
            int n = (ntBase + q) * 16 + (lane & 15);
            float bv = bias ? bias[n] : 0.f;
#pragma unroll
            for (int g = 0; g < 8; ++g) {
                int row = row0 + mb + g;
                float v = acc[q][g] + bv;
                if (act == 1)      v = silu_f(v);
                else if (act == 2) v = fmaxf(v, 0.f);
                if (resid) v += resid[(size_t)row * 256 + n];
                if (maskv) v *= maskv[row];
                if (outF) outF[(size_t)row * outStride + n] = v;
                if (outH) outH[(size_t)row * Npack + n] = (f16)v;
            }
        }
    } else {
        for (int t = wave; t < 2 * NT; t += 8) {
            int mt = t / NT, nt = t - mt * NT;
            v8f c = {};
            for (int kt = 0; kt < KT; ++kt) {
                v16h a  = lds_a_frag(Asm, mt, kt, lane, ld);
                v16h bm = Bv[(size_t)(kt * NT + nt) * 32 + lane];
                c = __builtin_amdgcn_wmma_f32_16x16x32_f16(false, a, false, bm, (short)0, c, false, false);
            }
            int n  = nt * 16 + (lane & 15);
            int mb = mt * 16 + (lane >> 4) * 8;
            float bv = (bias && n < Nwrite) ? bias[n] : 0.f;
#pragma unroll
            for (int g = 0; g < 8; ++g) {
                int row = row0 + mb + g;
                float v = c[g] + bv;
                if (act == 1)      v = silu_f(v);
                else if (act == 2) v = fmaxf(v, 0.f);
                if (resid) v += resid[(size_t)row * 256 + n];
                if (maskv) v *= maskv[row];
                if (n < Nwrite) {
                    if (outF) outF[(size_t)row * outStride + n] = v;
                    if (outH) outH[(size_t)row * Npack + n] = (f16)v;
                }
            }
        }
    }
}

// ---- shared pair-GEMM: Bout = silu([32,256] @ W2f + b2) -----------------
__device__ __forceinline__ void pair_gemm_silu(const f16* Asm, f16* Bout,
                                               const f16* __restrict__ W2f,
                                               const float* __restrict__ b2,
                                               int lane, int wave) {
    const v16h* Bv = reinterpret_cast<const v16h*>(W2f);
    const int mt = wave >> 2;             // 0..1
    const int ntBase = (wave & 3) * 4;    // 0,4,8,12
    v8f c0 = {}, c1 = {}, c2 = {}, c3 = {};
#pragma unroll
    for (int kt = 0; kt < 8; ++kt) {
        v16h a = lds_a_frag(Asm, mt, kt, lane, 264);
        const v16h* bp = Bv + (size_t)(kt * 16 + ntBase) * 32 + lane;
        v16h b0 = bp[0];
        v16h b1 = bp[32];
        v16h b2v = bp[64];
        v16h b3 = bp[96];
        c0 = __builtin_amdgcn_wmma_f32_16x16x32_f16(false, a, false, b0,  (short)0, c0, false, false);
        c1 = __builtin_amdgcn_wmma_f32_16x16x32_f16(false, a, false, b1,  (short)0, c1, false, false);
        c2 = __builtin_amdgcn_wmma_f32_16x16x32_f16(false, a, false, b2v, (short)0, c2, false, false);
        c3 = __builtin_amdgcn_wmma_f32_16x16x32_f16(false, a, false, b3,  (short)0, c3, false, false);
    }
    const int mb = mt * 16 + (lane >> 4) * 8;
    v8f acc[4] = {c0, c1, c2, c3};
#pragma unroll
    for (int q = 0; q < 4; ++q) {
        int n = (ntBase + q) * 16 + (lane & 15);
        float bv = b2[n];
#pragma unroll
        for (int g = 0; g < 8; ++g)
            Bout[(mb + g) * 264 + n] = (f16)silu_f(acc[q][g] + bv);
    }
}

// ---- fused pairwise edge branch: agg[b,i,:] -----------------------------
__global__ void pair_edge(const float* __restrict__ xcur, const float* __restrict__ coors,
                          const float* __restrict__ U, const float* __restrict__ V,
                          const float* __restrict__ wa, const float* __restrict__ wb,
                          const f16* __restrict__ W2f, const float* __restrict__ b2,
                          const float* __restrict__ attw, const float* __restrict__ attb,
                          const float* __restrict__ maskv, float* __restrict__ agg) {
    const int i = blockIdx.x, b = blockIdx.y;
    const int bi = b * Nn + i;
    const int tid = threadIdx.x;
    __shared__ alignas(32) f16 Asm[32 * 264];
    __shared__ alignas(32) f16 Bout[32 * 264];
    __shared__ float distL[Nn], dist0L[Nn];
    __shared__ float UiL[Hd], waL[Hd], wbL[Hd], vwL[Hd];
    __shared__ float coefL[32];

    UiL[tid] = U[(size_t)bi * Hd + tid];
    waL[tid] = wa[tid]; wbL[tid] = wb[tid]; vwL[tid] = attw[tid];
    if (tid < Nn) {
        int bj = b * Nn + tid;
        float d0 = xcur[bi * 3 + 0] - xcur[bj * 3 + 0];
        float d1 = xcur[bi * 3 + 1] - xcur[bj * 3 + 1];
        float d2 = xcur[bi * 3 + 2] - xcur[bj * 3 + 2];
        distL[tid] = d0 * d0 + d1 * d1 + d2 * d2;
        float e0 = coors[bi * 3 + 0] - coors[bj * 3 + 0];
        float e1 = coors[bi * 3 + 1] - coors[bj * 3 + 1];
        float e2 = coors[bi * 3 + 2] - coors[bj * 3 + 2];
        dist0L[tid] = e0 * e0 + e1 * e1 + e2 * e2;
    }
    __syncthreads();
    const float mi = maskv[bi];
    float aggAcc = 0.f;
    const int lane = tid & 31, wave = tid >> 5;

    for (int jc = 0; jc < 4; ++jc) {
        const int jb = jc * 32;
        {   // build silu(edge1) rows: col = tid
            float ui = UiL[tid], wan = waL[tid], wbn = wbL[tid];
            for (int r = 0; r < 32; ++r) {
                int j = jb + r;
                float v = ui + V[(size_t)(b * Nn + j) * Hd + tid]
                        + distL[j] * wan + dist0L[j] * wbn;
                Asm[r * 264 + tid] = (f16)silu_f(v);
            }
        }
        __syncthreads();
        pair_gemm_silu(Asm, Bout, W2f, b2, lane, wave);
        __syncthreads();
        // attention gate per row
        if (tid < 32) {
            int j = jb + tid;
            float dot = attb[0];
            for (int n = 0; n < Hd; ++n) dot += (float)Bout[tid * 264 + n] * vwL[n];
            float gate = 1.f / (1.f + __expf(-dot));
            float em = (j == i) ? 0.f : mi * maskv[b * Nn + j];
            coefL[tid] = gate * em;
        }
        __syncthreads();
        // masked scatter-sum over neighbors (col = tid)
        for (int r = 0; r < 32; ++r) aggAcc += (float)Bout[r * 264 + tid] * coefL[r];
        __syncthreads();
    }
    agg[(size_t)bi * Hd + tid] = aggAcc;
}

// ---- fused pairwise coordinate branch: xnext = (x + dx)*mask ------------
__global__ void pair_coord(const float* __restrict__ xcur, const float* __restrict__ coors,
                           const float* __restrict__ U, const float* __restrict__ V,
                           const float* __restrict__ wa, const float* __restrict__ wb,
                           const f16* __restrict__ W2f, const float* __restrict__ b2,
                           const float* __restrict__ c3w, const float* __restrict__ maskv,
                           float* __restrict__ xnext) {
    const int i = blockIdx.x, b = blockIdx.y;
    const int bi = b * Nn + i;
    const int tid = threadIdx.x;
    __shared__ alignas(32) f16 Asm[32 * 264];
    __shared__ alignas(32) f16 Bout[32 * 264];
    __shared__ float distL[Nn], dist0L[Nn], cdiffL[Nn * 3];
    __shared__ float UiL[Hd], waL[Hd], wbL[Hd], vwL[Hd];
    __shared__ float coefL[32];

    UiL[tid] = U[(size_t)bi * Hd + tid];
    waL[tid] = wa[tid]; wbL[tid] = wb[tid]; vwL[tid] = c3w[tid];
    if (tid < Nn) {
        int bj = b * Nn + tid;
        float d0 = xcur[bi * 3 + 0] - xcur[bj * 3 + 0];
        float d1 = xcur[bi * 3 + 1] - xcur[bj * 3 + 1];
        float d2 = xcur[bi * 3 + 2] - xcur[bj * 3 + 2];
        float r = d0 * d0 + d1 * d1 + d2 * d2;
        distL[tid] = r;
        float inv = 1.f / (sqrtf(r + 1e-8f) + 1.f);
        cdiffL[tid * 3 + 0] = d0 * inv;
        cdiffL[tid * 3 + 1] = d1 * inv;
        cdiffL[tid * 3 + 2] = d2 * inv;
        float e0 = coors[bi * 3 + 0] - coors[bj * 3 + 0];
        float e1 = coors[bi * 3 + 1] - coors[bj * 3 + 1];
        float e2 = coors[bi * 3 + 2] - coors[bj * 3 + 2];
        dist0L[tid] = e0 * e0 + e1 * e1 + e2 * e2;
    }
    __syncthreads();
    const float mi = maskv[bi];
    float dxAcc = 0.f;
    const int lane = tid & 31, wave = tid >> 5;

    for (int jc = 0; jc < 4; ++jc) {
        const int jb = jc * 32;
        {
            float ui = UiL[tid], wan = waL[tid], wbn = wbL[tid];
            for (int r = 0; r < 32; ++r) {
                int j = jb + r;
                float v = ui + V[(size_t)(b * Nn + j) * Hd + tid]
                        + distL[j] * wan + dist0L[j] * wbn;
                Asm[r * 264 + tid] = (f16)silu_f(v);
            }
        }
        __syncthreads();
        pair_gemm_silu(Asm, Bout, W2f, b2, lane, wave);
        __syncthreads();
        if (tid < 32) {
            int j = jb + tid;
            float dot = 0.f;
            for (int n = 0; n < Hd; ++n) dot += (float)Bout[tid * 264 + n] * vwL[n];
            float phi = tanhf(dot) * CRconst;
            float em = (j == i) ? 0.f : mi * maskv[b * Nn + j];
            coefL[tid] = phi * em;
        }
        __syncthreads();
        if (tid < 3) {
            for (int r = 0; r < 32; ++r)
                dxAcc += cdiffL[(jb + r) * 3 + tid] * coefL[r];
        }
        __syncthreads();
    }
    if (tid < 3) xnext[bi * 3 + tid] = (xcur[bi * 3 + tid] + dxAcc) * mi;
}

// ---- final mean removal per batch ---------------------------------------
__global__ void finalize_x(const float* __restrict__ xin, const float* __restrict__ maskv,
                           float* __restrict__ xout) {
    int b = blockIdx.x, t = threadIdx.x;      // 128 threads
    __shared__ float sx[Nn][3];
    __shared__ float sm[Nn];
    __shared__ float mean[3];
    int row = b * Nn + t;
    float m  = maskv[row];
    float x0 = xin[row * 3 + 0], x1 = xin[row * 3 + 1], x2 = xin[row * 3 + 2];
    sx[t][0] = x0; sx[t][1] = x1; sx[t][2] = x2; sm[t] = m;
    __syncthreads();
    for (int s = 64; s > 0; s >>= 1) {
        if (t < s) {
            sx[t][0] += sx[t + s][0]; sx[t][1] += sx[t + s][1]; sx[t][2] += sx[t + s][2];
            sm[t] += sm[t + s];
        }
        __syncthreads();
    }
    if (t == 0) {
        float nv = sm[0];
        mean[0] = sx[0][0] / nv; mean[1] = sx[0][1] / nv; mean[2] = sx[0][2] / nv;
    }
    __syncthreads();
    xout[row * 3 + 0] = x0 - mean[0] * m;
    xout[row * 3 + 1] = x1 - mean[1] * m;
    xout[row * 3 + 2] = x2 - mean[2] * m;
}

// ==========================================================================
extern "C" void kernel_launch(void* const* d_in, const int* in_sizes, int n_in,
                              void* d_out, int out_size, void* d_ws, size_t ws_size,
                              hipStream_t stream) {
    (void)in_sizes; (void)n_in; (void)out_size; (void)ws_size;
    // ---- inputs (jax pytree: dict keys sorted) ----
    const float* coors    = (const float*)d_in[0];
    const int*   atypes   = (const int*)d_in[1];
    const float* maskv    = (const float*)d_in[2];
    const int*   tC       = (const int*)d_in[3];
    const int*   tA       = (const int*)d_in[4];
    const float* atom_emb = (const float*)d_in[5];
    const float* cls1b = (const float*)d_in[6],  *cls1w = (const float*)d_in[7];
    const float* cls2b = (const float*)d_in[8],  *cls2w = (const float*)d_in[9];
    const float* cls3b = (const float*)d_in[10], *cls3w = (const float*)d_in[11];
    const float* embinb  = (const float*)d_in[12], *embinw  = (const float*)d_in[13];
    const float* emboutb = (const float*)d_in[14], *emboutw = (const float*)d_in[15];
    const float *attb[4], *attw_[4], *c1b[4], *c1w[4], *c2b[4], *c2w[4], *c3w_[4];
    const float *e1b[4], *e1w[4], *e2b[4], *e2w[4], *n1b[4], *n1w[4], *n2b[4], *n2w[4];
    for (int l = 0; l < 4; ++l) {
        int base = 16 + l * 15;
        attb[l]  = (const float*)d_in[base + 0];
        attw_[l] = (const float*)d_in[base + 1];
        c1b[l]   = (const float*)d_in[base + 2];
        c1w[l]   = (const float*)d_in[base + 3];
        c2b[l]   = (const float*)d_in[base + 4];
        c2w[l]   = (const float*)d_in[base + 5];
        c3w_[l]  = (const float*)d_in[base + 6];
        e1b[l]   = (const float*)d_in[base + 7];
        e1w[l]   = (const float*)d_in[base + 8];
        e2b[l]   = (const float*)d_in[base + 9];
        e2w[l]   = (const float*)d_in[base + 10];
        n1b[l]   = (const float*)d_in[base + 11];
        n1w[l]   = (const float*)d_in[base + 12];
        n2b[l]   = (const float*)d_in[base + 13];
        n2w[l]   = (const float*)d_in[base + 14];
    }
    const float* time_emb = (const float*)d_in[76];

    // ---- workspace carve-out ----
    char* ws = (char*)d_ws;
    size_t off = 0;
    auto alloc = [&](size_t bytes) -> void* {
        void* p = ws + off;
        off = (off + bytes + 255) & ~(size_t)255;
        return p;
    };
    const size_t Mrows = (size_t)Bsz * Nn;                 // 1024
    float* h    = (float*)alloc(Mrows * Hd * 4);
    float* e0   = (float*)alloc(Mrows * Hd * 4);
    float* U    = (float*)alloc(Mrows * Hd * 4);
    float* V    = (float*)alloc(Mrows * Hd * 4);
    float* agg  = (float*)alloc(Mrows * Hd * 4);
    float* gbuf = (float*)alloc(Mrows * Hd * 4);
    f16*   t1b  = (f16*)alloc(Mrows * Hd * 2);
    f16*   z1   = (f16*)alloc(Mrows * Hd * 2);
    f16*   z2   = (f16*)alloc(Mrows * Hd * 2);
    float* xa   = (float*)alloc(Mrows * 3 * 4);
    float* xb   = (float*)alloc(Mrows * 3 * 4);

    auto allocW = [&](int Kp, int Np) -> f16* { return (f16*)alloc((size_t)Kp * Np * 2); };
    auto pack = [&](const float* w, int Ksrc, int Nsrc, int Kp, int Np, f16* dst) {
        int total = Kp * Np;
        pack_wfrag<<<total / 256, 256, 0, stream>>>(w, Ksrc, Nsrc, Np / 16, dst, total);
    };

    f16 *W1a[4], *W1b[4], *W2[4], *Wc1a[4], *Wc1b[4], *Wc2[4], *Wn1[4], *Wn2[4];
    for (int l = 0; l < 4; ++l) {
        W1a[l]  = allocW(256, 256); pack(e1w[l],                256, 256, 256, 256, W1a[l]);
        W1b[l]  = allocW(256, 256); pack(e1w[l] + 256 * 256,    256, 256, 256, 256, W1b[l]);
        W2[l]   = allocW(256, 256); pack(e2w[l],                256, 256, 256, 256, W2[l]);
        Wc1a[l] = allocW(256, 256); pack(c1w[l],                256, 256, 256, 256, Wc1a[l]);
        Wc1b[l] = allocW(256, 256); pack(c1w[l] + 256 * 256,    256, 256, 256, 256, Wc1b[l]);
        Wc2[l]  = allocW(256, 256); pack(c2w[l],                256, 256, 256, 256, Wc2[l]);
        Wn1[l]  = allocW(512, 256); pack(n1w[l],                512, 256, 512, 256, Wn1[l]);
        Wn2[l]  = allocW(256, 256); pack(n2w[l],                256, 256, 256, 256, Wn2[l]);
    }
    f16* Fembin  = allocW(256, 256); pack(embinw,  256, 256, 256, 256, Fembin);
    f16* Fembout = allocW(256, 256); pack(emboutw, 256, 256, 256, 256, Fembout);
    f16* Fcls1   = allocW(256, 256); pack(cls1w,   256, 256, 256, 256, Fcls1);
    f16* Fcls2   = allocW(256, 256); pack(cls2w,   256, 256, 256, 256, Fcls2);
    f16* Fcls3   = allocW(256, 32);  pack(cls3w,   256, 21,  256, 32,  Fcls3);

    float* outX = (float*)d_out;
    float* outP = (float*)d_out + Mrows * 3;

    // ---- forward ----
    embed_kernel<<<dim3(Mrows), 256, 0, stream>>>(atom_emb, atypes, maskv, time_emb, tC, tA, e0);
    // h = emb_in(e0)
    node_gemm<<<32, 256, 0, stream>>>(e0, nullptr, nullptr, 256, 256, Fembin, embinb,
                                      256, 256, 0, nullptr, nullptr, h, 256, nullptr);

    const float* xc = coors;
    float* xn = xa;
    dim3 pairGrid(Nn, Bsz);
    for (int l = 0; l < 4; ++l) {
        // U/V for edge branch (edge1 decomposition; bias folded into U once)
        node_gemm<<<32, 256, 0, stream>>>(h, nullptr, nullptr, 256, 256, W1a[l], e1b[l],
                                          256, 256, 0, nullptr, nullptr, U, 256, nullptr);
        node_gemm<<<32, 256, 0, stream>>>(h, nullptr, nullptr, 256, 256, W1b[l], nullptr,
                                          256, 256, 0, nullptr, nullptr, V, 256, nullptr);
        pair_edge<<<pairGrid, 256, 0, stream>>>(xc, coors, U, V,
                                                e1w[l] + 512 * 256, e1w[l] + 513 * 256,
                                                W2[l], e2b[l], attw_[l], attb[l], maskv, agg);
        // node update: h = (h + node2(silu(node1([h, agg])))) * mask
        node_gemm<<<32, 256, 0, stream>>>(h, agg, nullptr, 256, 512, Wn1[l], n1b[l],
                                          256, 256, 1, nullptr, nullptr, nullptr, 0, t1b);
        node_gemm<<<32, 256, 0, stream>>>(nullptr, nullptr, t1b, 256, 256, Wn2[l], n2b[l],
                                          256, 256, 0, h, maskv, h, 256, nullptr);
        // U/V for coord branch (uses updated h)
        node_gemm<<<32, 256, 0, stream>>>(h, nullptr, nullptr, 256, 256, Wc1a[l], c1b[l],
                                          256, 256, 0, nullptr, nullptr, U, 256, nullptr);
        node_gemm<<<32, 256, 0, stream>>>(h, nullptr, nullptr, 256, 256, Wc1b[l], nullptr,
                                          256, 256, 0, nullptr, nullptr, V, 256, nullptr);
        pair_coord<<<pairGrid, 256, 0, stream>>>(xc, coors, U, V,
                                                 c1w[l] + 512 * 256, c1w[l] + 513 * 256,
                                                 Wc2[l], c2b[l], c3w_[l], maskv, xn);
        xc = xn;
        xn = (xn == xa) ? xb : xa;
    }

    // head: emb_out (masked) -> cls MLP -> pred
    node_gemm<<<32, 256, 0, stream>>>(h, nullptr, nullptr, 256, 256, Fembout, emboutb,
                                      256, 256, 0, nullptr, maskv, gbuf, 256, nullptr);
    node_gemm<<<32, 256, 0, stream>>>(gbuf, nullptr, nullptr, 256, 256, Fcls1, cls1b,
                                      256, 256, 2, nullptr, nullptr, nullptr, 0, z1);
    node_gemm<<<32, 256, 0, stream>>>(nullptr, nullptr, z1, 256, 256, Fcls2, cls2b,
                                      256, 256, 2, nullptr, nullptr, nullptr, 0, z2);
    node_gemm<<<32, 256, 0, stream>>>(nullptr, nullptr, z2, 256, 256, Fcls3, cls3b,
                                      32, Vv, 0, nullptr, nullptr, outP, Vv, nullptr);
    // x: remove mean with mask
    finalize_x<<<Bsz, Nn, 0, stream>>>(xc, maskv, outX);
}
